// EncoderModelLSH_26740466385573
// MI455X (gfx1250) — compile-verified, matrix operand order checked by reading, not run on hardware
//
#include <hip/hip_runtime.h>

typedef __attribute__((ext_vector_type(16))) _Float16 v16h;
typedef __attribute__((ext_vector_type(8)))  _Float16 v8h;
typedef __attribute__((ext_vector_type(8)))  float    v8f;

#define CB    4
#define CS    1024
#define CE    512
#define CH    8
#define CHD   64
#define CHID  2048
#define CNHYP 6
#define CNC   6

#define SC_STRIDE 1032   // 1024 + 8 floats padding (keeps 16B alignment)

// ---------------------------------------------------------------- WMMA helpers

__device__ __forceinline__ v8f wmma16(v16h a, v16h b, v8f c) {
  // v_wmma_f32_16x16x32_f16: D = A(16x32 f16) * B(32x16 f16) + C(16x16 f32)
  return __builtin_amdgcn_wmma_f32_16x16x32_f16(false, a, false, b, (short)0, c,
                                                false, false);
}

// A-operand (16x32, MxK) loader from row-major memory, also used for the
// B-operand when B is given as rows of W (C = A * W^T). Per ISA layout:
// lane l: m = l%16, half = l/16; halves hold K = {half*8..+7, 16+half*8..+7}.
__device__ __forceinline__ v16h load_row_frag(const _Float16* base, int ld) {
  const int lane = threadIdx.x & 31;
  const int m = lane & 15, half = lane >> 4;
  const _Float16* p = base + (size_t)m * ld + half * 8;
  v8h lo = *(const v8h*)(p);
  v8h hi = *(const v8h*)(p + 16);
  v16h r;
#pragma unroll
  for (int i = 0; i < 8; ++i) { r[i] = lo[i]; r[i + 8] = hi[i]; }
  return r;
}

// B-operand (32x16, KxN) loader from row-major KxN memory (column gather).
__device__ __forceinline__ v16h load_col_frag(const _Float16* base, int ld) {
  const int lane = threadIdx.x & 31;
  const int n = lane & 15, half = lane >> 4;
  v16h r;
#pragma unroll
  for (int v = 0; v < 8; ++v) {
    const int kb = ((v >> 2) << 4) + (half << 3) + ((v & 3) << 1);
    r[2 * v]     = base[(size_t)kb * ld + n];
    r[2 * v + 1] = base[(size_t)(kb + 1) * ld + n];
  }
  return r;
}

// A-operand built from f32 probabilities resident in LDS.
__device__ __forceinline__ v16h frag_from_lds(const float* sc, int r0, int k0) {
  const int lane = threadIdx.x & 31;
  const int m = lane & 15, half = lane >> 4;
  const float* p = sc + (size_t)(r0 + m) * SC_STRIDE + k0 + half * 8;
  v16h r;
#pragma unroll
  for (int i = 0; i < 8; ++i) {
    r[i]     = (_Float16)p[i];
    r[i + 8] = (_Float16)p[16 + i];
  }
  return r;
}

// ---------------------------------------------------------------- small utils

__global__ void cvt_f32_to_f16_kernel(const float* __restrict__ in,
                                      _Float16* __restrict__ out, int n) {
  int i = blockIdx.x * blockDim.x + threadIdx.x;
  if (i < n) out[i] = (_Float16)in[i];
}

// x = emb[inputs] + pe[batch]  (bug-faithful: PE indexed by batch)
__global__ void embed_pe_kernel(const int* __restrict__ inputs,
                                const float* __restrict__ emb,
                                float* __restrict__ x32,
                                _Float16* __restrict__ x16) {
  const int bs = blockIdx.x;               // b*S + s
  const int b = bs / CS;
  const int tok = inputs[bs];
  const float ln10k_over_E = 9.210340371976184f / (float)CE;
  for (int e = threadIdx.x; e < CE; e += blockDim.x) {
    const int i2 = (e >> 1) << 1;          // 2*(e/2)
    const float div = __expf(-(float)i2 * ln10k_over_E);
    const float ang = (float)b * div;
    const float pe = (e & 1) ? __cosf(ang) : __sinf(ang);
    const float v = emb[(size_t)tok * CE + e] + pe;
    x32[(size_t)bs * CE + e] = v;
    x16[(size_t)bs * CE + e] = (_Float16)v;
  }
}

// LSH bucket per (b,s,h) from q16
__global__ void bucket_kernel(const _Float16* __restrict__ q16,
                              const float* __restrict__ hyper,
                              int* __restrict__ buckets, int total) {
  int i = blockIdx.x * blockDim.x + threadIdx.x;
  if (i >= total) return;
  const int bs = i / CH, h = i % CH;
  const _Float16* qp = q16 + (size_t)bs * CE + h * CHD;
  int bucket = 0;
#pragma unroll
  for (int n = 0; n < CNHYP; ++n) {
    float acc = hyper[CHD * CNHYP + n];    // appended-1 bias hyperplane row
    for (int d = 0; d < CHD; ++d) acc += (float)qp[d] * hyper[d * CNHYP + n];
    if (acc >= 0.f) bucket |= (1 << n);
  }
  buckets[i] = bucket;
}

// LayerNorm per row (one wave / row), writes f32 + f16
__global__ __launch_bounds__(32)
void ln_kernel(const float* __restrict__ in, const float* __restrict__ g,
               const float* __restrict__ bvec, float* __restrict__ out32,
               _Float16* __restrict__ out16, int N) {
  const int row = blockIdx.x;
  const int lane = threadIdx.x;
  const float* p = in + (size_t)row * N;
  float s = 0.f, s2 = 0.f;
  for (int c = lane; c < N; c += 32) { float v = p[c]; s += v; s2 += v * v; }
#pragma unroll
  for (int o = 16; o > 0; o >>= 1) {
    s  += __shfl_xor(s,  o, 32);
    s2 += __shfl_xor(s2, o, 32);
  }
  const float mu = s / (float)N;
  const float var = s2 / (float)N - mu * mu;
  const float inv = rsqrtf(var + 1e-5f);
  for (int c = lane; c < N; c += 32) {
    const float v = (p[c] - mu) * inv * g[c] + bvec[c];
    out32[(size_t)row * N + c] = v;
    out16[(size_t)row * N + c] = (_Float16)v;
  }
}

// ---------------------------------------------------------------- WMMA GEMM
// C[M,N] = A[M,K](f16) * W[N,K](f16)^T  (+bias)(relu?)(+resid) -> out32/out16
// One wave per 32x32 output tile, 2x2 fragment blocking.
__global__ __launch_bounds__(32)
void gemm_xwt_kernel(const _Float16* __restrict__ A,
                     const _Float16* __restrict__ W,
                     const float* __restrict__ bias,
                     const float* __restrict__ resid,
                     float* __restrict__ out32, _Float16* __restrict__ out16,
                     int M, int N, int K, int relu) {
  const int tilesN = N >> 5;
  const int tm = blockIdx.x / tilesN;
  const int tn = blockIdx.x % tilesN;
  const int row0 = tm << 5, col0 = tn << 5;
  const int lane = threadIdx.x & 31;
  const int n = lane & 15, half = lane >> 4;

  v8f zz = {};
  v8f acc[2][2];
  acc[0][0] = zz; acc[0][1] = zz; acc[1][0] = zz; acc[1][1] = zz;

  for (int k0 = 0; k0 < K; k0 += 32) {
    v16h a0 = load_row_frag(A + (size_t)row0 * K + k0, K);
    v16h a1 = load_row_frag(A + (size_t)(row0 + 16) * K + k0, K);
    v16h b0 = load_row_frag(W + (size_t)col0 * K + k0, K);
    v16h b1 = load_row_frag(W + (size_t)(col0 + 16) * K + k0, K);
    acc[0][0] = wmma16(a0, b0, acc[0][0]);
    acc[0][1] = wmma16(a0, b1, acc[0][1]);
    acc[1][0] = wmma16(a1, b0, acc[1][0]);
    acc[1][1] = wmma16(a1, b1, acc[1][1]);
  }

#pragma unroll
  for (int i = 0; i < 2; ++i) {
#pragma unroll
    for (int j = 0; j < 2; ++j) {
      const int col = col0 + j * 16 + n;
#pragma unroll
      for (int r = 0; r < 8; ++r) {
        const int row = row0 + i * 16 + r + half * 8;
        float v = acc[i][j][r];
        if (bias)  v += bias[col];
        if (relu)  v = v > 0.f ? v : 0.f;
        if (resid) v += resid[(size_t)row * N + col];
        if (out32) out32[(size_t)row * N + col] = v;
        if (out16) out16[(size_t)row * N + col] = (_Float16)v;
      }
    }
  }
}

// ---------------------------------------------------------------- attention
// One 128-thread workgroup per (b, h, 32-row strip).
//   attn = softmax( (Q.Q^T)/sqrt(E) * (62 + same_bucket) ), out = attn @ V
__global__ __launch_bounds__(128)
void lsh_attn_kernel(const _Float16* __restrict__ q16,
                     const _Float16* __restrict__ v16p,
                     const int* __restrict__ buckets,
                     float* __restrict__ out32) {
  extern __shared__ float sc[];            // [32][SC_STRIDE] scores/probs
  __shared__ float redm[32][4];
  __shared__ float reds[32][4];

  const int sblk = blockIdx.x & 31;        // S/32 == 32 strips
  const int bh = blockIdx.x >> 5;
  const int bb = bh / CH;
  const int hh = bh % CH;
  const int s0 = sblk << 5;

  const int tid = threadIdx.x;
  const int wave = tid >> 5;
  const int lane = tid & 31;
  const int nloc = lane & 15, half = lane >> 4;

  const _Float16* qhead = q16 + (size_t)bb * CS * CE + hh * CHD;
  const _Float16* vhead = v16p + (size_t)bb * CS * CE + hh * CHD;
  const int* bk = buckets + (size_t)bb * CS * CH + hh;  // bucket(s) = bk[s*CH]

  const float scale = 0.044194173824159216f;            // 1/sqrt(512)

  // Phase 1: 32x1024 score strip via WMMA (Q is both A and B operand)
  v16h aQ[2][2];
#pragma unroll
  for (int i = 0; i < 2; ++i)
#pragma unroll
    for (int ks = 0; ks < 2; ++ks)
      aQ[i][ks] = load_row_frag(qhead + (size_t)(s0 + i * 16) * CE + ks * 32, CE);

  for (int it = 0; it < 8; ++it) {
    const int t0 = (wave + 4 * it) << 5;
    v8f zz = {};
    v8f acc[2][2];
    acc[0][0] = zz; acc[0][1] = zz; acc[1][0] = zz; acc[1][1] = zz;
#pragma unroll
    for (int ks = 0; ks < 2; ++ks) {
      v16h b0 = load_row_frag(qhead + (size_t)t0 * CE + ks * 32, CE);
      v16h b1 = load_row_frag(qhead + (size_t)(t0 + 16) * CE + ks * 32, CE);
      acc[0][0] = wmma16(aQ[0][ks], b0, acc[0][0]);
      acc[0][1] = wmma16(aQ[0][ks], b1, acc[0][1]);
      acc[1][0] = wmma16(aQ[1][ks], b0, acc[1][0]);
      acc[1][1] = wmma16(aQ[1][ks], b1, acc[1][1]);
    }
#pragma unroll
    for (int i = 0; i < 2; ++i) {
#pragma unroll
      for (int j = 0; j < 2; ++j) {
        const int t = t0 + j * 16 + nloc;
        const int bkt = bk[(size_t)t * CH];
#pragma unroll
        for (int r = 0; r < 8; ++r) {
          const int lr = i * 16 + r + half * 8;
          const int bks = bk[(size_t)(s0 + lr) * CH];
          const float mult = 62.0f + (bks == bkt ? 1.0f : 0.0f);
          sc[(size_t)lr * SC_STRIDE + t] = acc[i][j][r] * scale * mult;
        }
      }
    }
  }
  __syncthreads();

  // Phase 2: row softmax (4 threads per row, 256 cols each)
  {
    const int r = tid >> 2, sub = tid & 3;
    float* rowp = sc + (size_t)r * SC_STRIDE + sub * 256;
    float mx = -3.0e38f;
    for (int c = 0; c < 256; ++c) mx = fmaxf(mx, rowp[c]);
    redm[r][sub] = mx;
    __syncthreads();
    mx = fmaxf(fmaxf(redm[r][0], redm[r][1]), fmaxf(redm[r][2], redm[r][3]));
    float sum = 0.f;
    for (int c = 0; c < 256; ++c) {
      const float e = __expf(rowp[c] - mx);
      rowp[c] = e;
      sum += e;
    }
    reds[r][sub] = sum;
    __syncthreads();
    const float inv = 1.0f / (reds[r][0] + reds[r][1] + reds[r][2] + reds[r][3]);
    for (int c = 0; c < 256; ++c) rowp[c] *= inv;
  }
  __syncthreads();

  // Phase 3: out = P @ V; each wave owns a 16-wide d strip
  {
    const int d0 = wave << 4;
    v8f zz = {};
    v8f acc0 = zz, acc1 = zz;
    for (int k0 = 0; k0 < CS; k0 += 32) {
      v16h a0 = frag_from_lds(sc, 0, k0);
      v16h a1 = frag_from_lds(sc, 16, k0);
      v16h bf = load_col_frag(vhead + (size_t)k0 * CE + d0, CE);
      acc0 = wmma16(a0, bf, acc0);
      acc1 = wmma16(a1, bf, acc1);
    }
#pragma unroll
    for (int r = 0; r < 8; ++r) {
      const int m = r + half * 8;
      out32[((size_t)bb * CS + s0 + m)      * CE + hh * CHD + d0 + nloc] = acc0[r];
      out32[((size_t)bb * CS + s0 + 16 + m) * CE + hh * CHD + d0 + nloc] = acc1[r];
    }
  }
}

// ---------------------------------------------------------------- classifier
__global__ __launch_bounds__(256)
void out_head_kernel(const float* __restrict__ x32,
                     const float* __restrict__ Wout,
                     const float* __restrict__ bout, float* __restrict__ y) {
  const int b = blockIdx.x / CNC, c = blockIdx.x % CNC;
  const float* xp = x32 + (size_t)b * CS * CE;
  const float* wp = Wout + (size_t)c * CS * CE;
  float s = 0.f;
  for (int i = threadIdx.x; i < CS * CE; i += 256) s += xp[i] * wp[i];
  __shared__ float red[256];
  red[threadIdx.x] = s;
  __syncthreads();
  for (int o = 128; o > 0; o >>= 1) {
    if (threadIdx.x < o) red[threadIdx.x] += red[threadIdx.x + o];
    __syncthreads();
  }
  if (threadIdx.x == 0) y[blockIdx.x] = red[0] + bout[c];
}

// ---------------------------------------------------------------- launcher

extern "C" void kernel_launch(void* const* d_in, const int* in_sizes, int n_in,
                              void* d_out, int out_size, void* d_ws, size_t ws_size,
                              hipStream_t stream) {
  const int*   inputs = (const int*)d_in[0];
  const float* emb    = (const float*)d_in[1];
  const float* hyper  = (const float*)d_in[2];
  const float* Wq     = (const float*)d_in[3];
  const float* bq     = (const float*)d_in[4];
  const float* Wv     = (const float*)d_in[5];
  const float* bv     = (const float*)d_in[6];
  const float* ln_g   = (const float*)d_in[7];
  const float* ln_b   = (const float*)d_in[8];
  const float* W1     = (const float*)d_in[9];
  const float* b1     = (const float*)d_in[10];
  const float* W2     = (const float*)d_in[11];
  const float* b2     = (const float*)d_in[12];
  const float* Wout   = (const float*)d_in[13];
  const float* bout   = (const float*)d_in[14];
  float* y = (float*)d_out;

  const int M = CB * CS;                      // 4096 token rows

  char* ws = (char*)d_ws;
  size_t off = 0;
  auto alloc = [&](size_t bytes) -> void* {
    void* p = ws + off;
    off = (off + bytes + 255) & ~(size_t)255;
    return p;
  };
  float*    x32   = (float*)   alloc((size_t)M * CE * 4);
  _Float16* x16   = (_Float16*)alloc((size_t)M * CE * 2);
  _Float16* q16   = (_Float16*)alloc((size_t)M * CE * 2);
  _Float16* v16b  = (_Float16*)alloc((size_t)M * CE * 2);
  int*      bkts  = (int*)     alloc((size_t)M * CH * 4);
  float*    ao32  = (float*)   alloc((size_t)M * CE * 4);
  _Float16* h16   = (_Float16*)alloc((size_t)M * CHID * 2);
  float*    f32b  = (float*)   alloc((size_t)M * CE * 4);
  _Float16* Wq16  = (_Float16*)alloc((size_t)CE * CE * 2);
  _Float16* Wv16  = (_Float16*)alloc((size_t)CE * CE * 2);
  _Float16* W116  = (_Float16*)alloc((size_t)CHID * CE * 2);
  _Float16* W216  = (_Float16*)alloc((size_t)CE * CHID * 2);
  (void)ws_size; (void)n_in; (void)in_sizes; (void)out_size;

  const int smem_attn = 32 * SC_STRIDE * 4;   // 132 KB dynamic LDS (WGP has 320 KB)
  (void)hipFuncSetAttribute((const void*)lsh_attn_kernel,
                            hipFuncAttributeMaxDynamicSharedMemorySize, smem_attn);

  // Weight conversion (f32 -> f16), once per call (deterministic)
  {
    int n;
    n = CE * CE;
    cvt_f32_to_f16_kernel<<<(n + 255) / 256, 256, 0, stream>>>(Wq, Wq16, n);
    cvt_f32_to_f16_kernel<<<(n + 255) / 256, 256, 0, stream>>>(Wv, Wv16, n);
    n = CHID * CE;
    cvt_f32_to_f16_kernel<<<(n + 255) / 256, 256, 0, stream>>>(W1, W116, n);
    cvt_f32_to_f16_kernel<<<(n + 255) / 256, 256, 0, stream>>>(W2, W216, n);
  }

  // Embedding + positional encoding
  embed_pe_kernel<<<M, 256, 0, stream>>>(inputs, emb, x32, x16);

  for (int layer = 0; layer < 6; ++layer) {
    // Q and V projections (f16 out only)
    gemm_xwt_kernel<<<(M / 32) * (CE / 32), 32, 0, stream>>>(
        x16, Wq16, bq, nullptr, nullptr, q16, M, CE, CE, 0);
    gemm_xwt_kernel<<<(M / 32) * (CE / 32), 32, 0, stream>>>(
        x16, Wv16, bv, nullptr, nullptr, v16b, M, CE, CE, 0);

    // LSH buckets
    bucket_kernel<<<(M * CH + 255) / 256, 256, 0, stream>>>(
        q16, hyper, bkts, M * CH);

    // Attention (scores + bucket-weighted softmax + P@V)
    lsh_attn_kernel<<<CB * CH * (CS / 32), 128, smem_attn, stream>>>(
        q16, v16b, bkts, ao32);

    // LN #1
    ln_kernel<<<M, 32, 0, stream>>>(ao32, ln_g, ln_b, x32, x16, CE);

    // FFN: relu(x W1^T + b1) -> h16 ; (h W2^T + b2) + x -> f32b ; LN #2
    gemm_xwt_kernel<<<(M / 32) * (CHID / 32), 32, 0, stream>>>(
        x16, W116, b1, nullptr, nullptr, h16, M, CHID, CE, 1);
    gemm_xwt_kernel<<<(M / 32) * (CE / 32), 32, 0, stream>>>(
        h16, W216, b2, x32, f32b, nullptr, M, CE, CHID, 0);
    ln_kernel<<<M, 32, 0, stream>>>(f32b, ln_g, ln_b, x32, x16, CE);
  }

  // Classifier head: y[b,c] = x.flatten(b) . Wout[c] + bout[c]
  out_head_kernel<<<CB * CNC, 256, 0, stream>>>(x32, Wout, bout, y);
}
